// EdgeCentricRGCN_86294482911410
// MI455X (gfx1250) — matrix operand order repaired and checked by reading, not compiled, not run on hardware
//
#include <hip/hip_runtime.h>
#include <hip/hip_bf16.h>
#include <math.h>

#define N_NODES 50000
#define N_EDGES 800000
#define HIDDEN  128
#define N_GRAPHS 64

typedef __attribute__((ext_vector_type(16))) _Float16 v16h;
typedef __attribute__((ext_vector_type(8)))  _Float16 v8h;
typedef __attribute__((ext_vector_type(8)))  float    v8f;

// Relaxed, device-scope f32 atomic add (no return) -> should select
// GLOBAL_ATOMIC_ADD_F32 without return (STOREcnt path) instead of a CAS loop.
__device__ __forceinline__ void atomic_add_f32_fast(float* p, float v) {
    (void)__hip_atomic_fetch_add(p, v, __ATOMIC_RELAXED, __HIP_MEMORY_SCOPE_AGENT);
}

// ---------------------------------------------------------------------------
// Pack a [128(K) x 128(N)] row-major f32 weight into WMMA B-fragment order:
// wpack[((kb*8+nb)*32 + lane)*16 + i] = (f16) W[kb*32 + (lane>>4)*8 + i + (i>=8 ? 8 : 0)][nb*16 + (lane&15)]
// so each wave's B fragment is one contiguous 32B v16h load per lane.
// ---------------------------------------------------------------------------
__global__ void pack_w_kernel(const float* __restrict__ w0, const float* __restrict__ w1,
                              const float* __restrict__ w2, const float* __restrict__ w3,
                              _Float16* __restrict__ p0, _Float16* __restrict__ p1,
                              _Float16* __restrict__ p2, _Float16* __restrict__ p3) {
    int kb = blockIdx.x >> 3;
    int nb = blockIdx.x & 7;
    int lane = threadIdx.x;            // 0..31
    const float* w; _Float16* p;
    switch (blockIdx.y) {
        case 0:  w = w0; p = p0; break;
        case 1:  w = w1; p = p1; break;
        case 2:  w = w2; p = p2; break;
        default: w = w3; p = p3; break;
    }
    int base = ((kb * 8 + nb) * 32 + lane) * 16;
    int krow = kb * 32 + ((lane >> 4) * 8);
    int col  = nb * 16 + (lane & 15);
#pragma unroll
    for (int i = 0; i < 16; ++i)
        p[base + i] = (_Float16)w[(krow + i + (i >= 8 ? 8 : 0)) * HIDDEN + col];
}

// Fold BN(eval) into per-channel scale/shift for both layers.
__global__ void bn_prep_kernel(const float* g1, const float* b1, const float* m1, const float* v1,
                               const float* g2, const float* b2, const float* m2, const float* v2,
                               float* sc1, float* sh1, float* sc2, float* sh2) {
    int i = threadIdx.x;
    if (i < HIDDEN) {
        float s = g1[i] * rsqrtf(v1[i] + 1e-5f);
        sc1[i] = s; sh1[i] = b1[i] - m1[i] * s;
    } else {
        int j = i - HIDDEN;
        float s = g2[j] * rsqrtf(v2[j] + 1e-5f);
        sc2[j] = s; sh2[j] = b2[j] - m2[j] * s;
    }
}

// Node encoder: x[n,c] = node_weight[n]*ne_w[c] + ne_b[c]
__global__ void encode_kernel(const float* __restrict__ nw, const float* __restrict__ ew,
                              const float* __restrict__ eb, float* __restrict__ x) {
    int tid = blockIdx.x * 256 + threadIdx.x;       // < N_NODES*128
    int n = tid >> 7, c = tid & 127;
    x[tid] = nw[n] * ew[c] + eb[c];
}

// Edge message + scatter-add: agg[dst] += relu(x[src] + ea),  ea computed on the fly (rank-1).
__global__ void edge_kernel(const float* __restrict__ x,
                            const int* __restrict__ src, const int* __restrict__ dst,
                            const float* __restrict__ ewt, const float* __restrict__ eew,
                            const float* __restrict__ eeb, float* __restrict__ agg) {
    int tid = blockIdx.x * 256 + threadIdx.x;       // < N_EDGES*128
    int e = tid >> 7, c = tid & 127;
    float ea = ewt[e] * eew[c] + eeb[c];
    float m  = x[(size_t)src[e] * HIDDEN + c] + ea;
    m = fmaxf(m, 0.0f);
    atomic_add_f32_fast(&agg[(size_t)dst[e] * HIDDEN + c], m);
}

// h16 = f16( (1+eps)*x + agg )  -- input activations for the WMMA MLP
__global__ void node_upd_kernel(const float* __restrict__ x, const float* __restrict__ agg,
                                const float* __restrict__ eps, _Float16* __restrict__ h16) {
    int tid = blockIdx.x * 256 + threadIdx.x;       // < N_NODES*128
    h16[tid] = (_Float16)((1.0f + eps[0]) * x[tid] + agg[tid]);
}

// ---------------------------------------------------------------------------
// Fused GINE MLP: xout = relu( bn_affine( leaky(h16 @ W1 + b1) @ W2 + b2 ) )
// One workgroup (8 waves) per 16-row tile. Wave w owns N-tile w in both GEMMs.
// Stage-1 output staged in LDS (f16, 16x128). 8x v_wmma_f32_16x16x32_f16 / wave.
// ---------------------------------------------------------------------------
__global__ void __launch_bounds__(256)
gine_mlp_kernel(const _Float16* __restrict__ h16,
                const _Float16* __restrict__ w1p, const float* __restrict__ b1,
                const _Float16* __restrict__ w2p, const float* __restrict__ b2,
                const float* __restrict__ sc, const float* __restrict__ sh,
                float* __restrict__ xout) {
    __shared__ _Float16 tbuf[16 * HIDDEN];
    const int lane = threadIdx.x & 31;
    const int w    = threadIdx.x >> 5;        // N-tile 0..7
    const int m0   = blockIdx.x * 16;
    const int lrow = lane & 15;
    const int koff = (lane >> 4) * 8;         // A-frag K sub-offset (ISA 16-bit A layout)
    const int col  = w * 16 + lrow;           // D column for this lane

    // ---- stage 1: t = leaky(h0 @ W1 + b1) ----
    v8f acc = {};
#pragma unroll
    for (int kb = 0; kb < 4; ++kb) {
        int kbase = kb * 32 + koff;
        const v8h lo = *(const v8h*)(h16 + (size_t)(m0 + lrow) * HIDDEN + kbase);
        const v8h hi = *(const v8h*)(h16 + (size_t)(m0 + lrow) * HIDDEN + kbase + 16);
        v16h a;
#pragma unroll
        for (int i = 0; i < 8; ++i) { a[i] = lo[i]; a[8 + i] = hi[i]; }
        v16h bm = *(const v16h*)(w1p + (size_t)((kb * 8 + w) * 32 + lane) * 16);
        acc = __builtin_amdgcn_wmma_f32_16x16x32_f16(false, a, false, bm,
                                                     (short)0, acc, false, false);
    }
    {
        float bias = b1[col];
#pragma unroll
        for (int r = 0; r < 8; ++r) {
            float v = acc[r] + bias;
            v = (v >= 0.0f) ? v : 0.01f * v;          // leaky relu
            int row = r + ((lane >> 4) << 3);          // D row layout
            tbuf[row * HIDDEN + col] = (_Float16)v;
        }
    }
    __syncthreads();

    // ---- stage 2: y = t @ W2 + b2; bn affine; relu ----
    v8f acc2 = {};
#pragma unroll
    for (int kb = 0; kb < 4; ++kb) {
        int kbase = kb * 32 + koff;
        const v8h lo = *(const v8h*)(tbuf + lrow * HIDDEN + kbase);
        const v8h hi = *(const v8h*)(tbuf + lrow * HIDDEN + kbase + 16);
        v16h a;
#pragma unroll
        for (int i = 0; i < 8; ++i) { a[i] = lo[i]; a[8 + i] = hi[i]; }
        v16h bm = *(const v16h*)(w2p + (size_t)((kb * 8 + w) * 32 + lane) * 16);
        acc2 = __builtin_amdgcn_wmma_f32_16x16x32_f16(false, a, false, bm,
                                                      (short)0, acc2, false, false);
    }
    {
        float b2v = b2[col], s = sc[col], t = sh[col];
#pragma unroll
        for (int r = 0; r < 8; ++r) {
            float v = (acc2[r] + b2v) * s + t;
            v = fmaxf(v, 0.0f);
            xout[(size_t)(m0 + r + ((lane >> 4) << 3)) * HIDDEN + col] = v;
        }
    }
}

// Global mean-pool (sums + counts via atomics; pooled/cnt zeroed beforehand).
__global__ void pool_kernel(const float* __restrict__ x, const int* __restrict__ batch,
                            float* __restrict__ pooled, float* __restrict__ cnt) {
    int tid = blockIdx.x * 256 + threadIdx.x;       // < N_NODES*128
    int n = tid >> 7, c = tid & 127;
    int g = batch[n];
    atomic_add_f32_fast(&pooled[g * HIDDEN + c], x[tid]);
    if (c == 0) atomic_add_f32_fast(&cnt[g], 1.0f);
}

// Head: out[g] = sigmoid( leaky(mean_g @ W1 + b1) @ W2 + b2 )
__global__ void head_kernel(const float* __restrict__ pooled, const float* __restrict__ cnt,
                            const float* __restrict__ w1, const float* __restrict__ hb1,
                            const float* __restrict__ w2, const float* __restrict__ hb2,
                            float* __restrict__ out) {
    __shared__ float p[HIDDEN];
    __shared__ float red[HIDDEN];
    int g = blockIdx.x, t = threadIdx.x;
    float c = cnt[g]; c = (c > 1.0f) ? c : 1.0f;
    p[t] = pooled[g * HIDDEN + t] / c;
    __syncthreads();
    float h = hb1[t];
#pragma unroll 8
    for (int k = 0; k < HIDDEN; ++k) h += p[k] * w1[k * HIDDEN + t];
    h = (h >= 0.0f) ? h : 0.01f * h;
    red[t] = h * w2[t];
    __syncthreads();
    for (int s = 64; s > 0; s >>= 1) {
        if (t < s) red[t] += red[t + s];
        __syncthreads();
    }
    if (t == 0) {
        float z = red[0] + hb2[0];
        out[g] = 1.0f / (1.0f + expf(-z));
    }
}

// ---------------------------------------------------------------------------
extern "C" void kernel_launch(void* const* d_in, const int* in_sizes, int n_in,
                              void* d_out, int out_size, void* d_ws, size_t ws_size,
                              hipStream_t stream) {
    (void)in_sizes; (void)n_in; (void)out_size; (void)ws_size;

    const float* node_w = (const float*)d_in[0];
    const float* edge_w = (const float*)d_in[1];
    const int*   eidx   = (const int*)d_in[2];
    const int*   batch  = (const int*)d_in[3];
    const float* ne_w   = (const float*)d_in[4];
    const float* ne_b   = (const float*)d_in[5];
    const float* ee_w   = (const float*)d_in[6];
    const float* ee_b   = (const float*)d_in[7];
    const float* eps1   = (const float*)d_in[8];
    const float* m1w1   = (const float*)d_in[9];
    const float* m1b1   = (const float*)d_in[10];
    const float* m1w2   = (const float*)d_in[11];
    const float* m1b2   = (const float*)d_in[12];
    const float* bn1g   = (const float*)d_in[13];
    const float* bn1b   = (const float*)d_in[14];
    const float* bn1m   = (const float*)d_in[15];
    const float* bn1v   = (const float*)d_in[16];
    const float* eps2   = (const float*)d_in[17];
    const float* m2w1   = (const float*)d_in[18];
    const float* m2b1   = (const float*)d_in[19];
    const float* m2w2   = (const float*)d_in[20];
    const float* m2b2   = (const float*)d_in[21];
    const float* bn2g   = (const float*)d_in[22];
    const float* bn2b   = (const float*)d_in[23];
    const float* bn2m   = (const float*)d_in[24];
    const float* bn2v   = (const float*)d_in[25];
    const float* hw1    = (const float*)d_in[26];
    const float* hb1    = (const float*)d_in[27];
    const float* hw2    = (const float*)d_in[28];
    const float* hb2    = (const float*)d_in[29];

    const int* src = eidx;
    const int* dst = eidx + N_EDGES;

    // --- workspace carve-out (256B aligned) ---
    char* base = (char*)d_ws;
    size_t off = 0;
    auto carve = [&](size_t bytes) -> char* {
        char* p = base + off;
        off = (off + bytes + 255) & ~(size_t)255;
        return p;
    };
    const size_t xb = (size_t)N_NODES * HIDDEN * sizeof(float);
    float*     x    = (float*)carve(xb);
    float*     agg  = (float*)carve(xb);
    _Float16*  h16  = (_Float16*)carve((size_t)N_NODES * HIDDEN * sizeof(_Float16));
    _Float16*  wp0  = (_Float16*)carve(HIDDEN * HIDDEN * sizeof(_Float16)); // mlp1_w1
    _Float16*  wp1  = (_Float16*)carve(HIDDEN * HIDDEN * sizeof(_Float16)); // mlp1_w2
    _Float16*  wp2  = (_Float16*)carve(HIDDEN * HIDDEN * sizeof(_Float16)); // mlp2_w1
    _Float16*  wp3  = (_Float16*)carve(HIDDEN * HIDDEN * sizeof(_Float16)); // mlp2_w2
    float*     sc1  = (float*)carve(HIDDEN * sizeof(float));
    float*     sh1  = (float*)carve(HIDDEN * sizeof(float));
    float*     sc2  = (float*)carve(HIDDEN * sizeof(float));
    float*     sh2  = (float*)carve(HIDDEN * sizeof(float));
    float*     pooled = (float*)carve((N_GRAPHS * HIDDEN + N_GRAPHS) * sizeof(float));
    float*     cnt  = pooled + N_GRAPHS * HIDDEN;

    const int nodeBlocks = (N_NODES * HIDDEN) / 256;  // 25000
    const int edgeBlocks = (N_EDGES * HIDDEN) / 256;  // 400000
    const int mTiles     = N_NODES / 16;              // 3125

    // --- prep ---
    pack_w_kernel<<<dim3(32, 4), 32, 0, stream>>>(m1w1, m1w2, m2w1, m2w2, wp0, wp1, wp2, wp3);
    bn_prep_kernel<<<1, 256, 0, stream>>>(bn1g, bn1b, bn1m, bn1v, bn2g, bn2b, bn2m, bn2v,
                                          sc1, sh1, sc2, sh2);
    encode_kernel<<<nodeBlocks, 256, 0, stream>>>(node_w, ne_w, ne_b, x);

    // --- conv 1 ---
    hipMemsetAsync(agg, 0, xb, stream);
    edge_kernel<<<edgeBlocks, 256, 0, stream>>>(x, src, dst, edge_w, ee_w, ee_b, agg);
    node_upd_kernel<<<nodeBlocks, 256, 0, stream>>>(x, agg, eps1, h16);
    gine_mlp_kernel<<<mTiles, 256, 0, stream>>>(h16, wp0, m1b1, wp1, m1b2, sc1, sh1, x);

    // --- conv 2 ---
    hipMemsetAsync(agg, 0, xb, stream);
    edge_kernel<<<edgeBlocks, 256, 0, stream>>>(x, src, dst, edge_w, ee_w, ee_b, agg);
    node_upd_kernel<<<nodeBlocks, 256, 0, stream>>>(x, agg, eps2, h16);
    gine_mlp_kernel<<<mTiles, 256, 0, stream>>>(h16, wp2, m2b1, wp3, m2b2, sc2, sh2, x);

    // --- pool + head ---
    hipMemsetAsync(pooled, 0, (N_GRAPHS * HIDDEN + N_GRAPHS) * sizeof(float), stream);
    pool_kernel<<<nodeBlocks, 256, 0, stream>>>(x, batch, pooled, cnt);
    head_kernel<<<N_GRAPHS, HIDDEN, 0, stream>>>(pooled, cnt, hw1, hb1, hw2, hb2, (float*)d_out);
}
